// MSA_78451872628976
// MI455X (gfx1250) — matrix operand order repaired and checked by reading, not compiled, not run on hardware
//
#include <hip/hip_runtime.h>
#include <hip/hip_bf16.h>
#include <math.h>

// ---------------------------------------------------------------------------
// Types & helpers for CDNA5 WMMA (gfx1250, wave32)
// ---------------------------------------------------------------------------
typedef __bf16 bf16_t;
typedef __attribute__((ext_vector_type(16))) __bf16 v16bf;
typedef __attribute__((ext_vector_type(8)))  float  v8f;

union BFrag { v16bf v; uint4 q[2]; bf16_t h[16]; };

__device__ __forceinline__ bf16_t f2bf(float f) {
  unsigned u = __builtin_bit_cast(unsigned, f);
  unsigned r = (u + 0x7FFFu + ((u >> 16) & 1u)) >> 16;   // round-to-nearest-even
  unsigned short s = (unsigned short)r;
  return __builtin_bit_cast(bf16_t, s);
}
__device__ __forceinline__ float bf2f(bf16_t h) {
  unsigned short s = __builtin_bit_cast(unsigned short, h);
  unsigned u = ((unsigned)s) << 16;
  return __builtin_bit_cast(float, u);
}

// CDNA5 async global->LDS copy (16B per lane), tracked by ASYNCcnt.
// lds_off = wave-relative LDS byte address (low 32 bits of generic address).
__device__ __forceinline__ void async_ld16(unsigned lds_off, const void* gptr) {
  asm volatile("global_load_async_to_lds_b128 %0, %1, off"
               :: "v"(lds_off), "v"(gptr) : "memory");
}
template <int N>
__device__ __forceinline__ void wait_async() {
  asm volatile("s_wait_asynccnt %0" :: "i"(N) : "memory");
}
__device__ __forceinline__ unsigned lds_addr(const void* p) {
  return (unsigned)(unsigned long long)p;
}

// A-matrix fragment (16x32 bf16, MxK), row-major source with row stride rs.
// lane L holds row m0+(L&15); elems j -> k = half*16 + j (contiguous,
// half = L>>4), matching load_frag_b. Two b128 loads.
__device__ __forceinline__ v16bf load_frag_a(const bf16_t* base, int rs, int m0, int k0) {
  int lane = threadIdx.x & 31;
  int m = m0 + (lane & 15);
  int half = lane >> 4;
  const bf16_t* p = base + (size_t)m * rs + k0 + half * 16;
  BFrag f;
  f.q[0] = *(const uint4*)(p);
  f.q[1] = *(const uint4*)(p + 8);
  return f.v;
}

// B-matrix fragment (32x16 bf16, KxN) from a [N][K] row-major source.
// lane L holds col n0+(L&15); elems j -> k = (L>>4)*16 + j (contiguous).
__device__ __forceinline__ v16bf load_frag_b(const bf16_t* base, int rs, int n0, int k0) {
  int lane = threadIdx.x & 31;
  int n = n0 + (lane & 15);
  int kb = k0 + (lane >> 4) * 16;
  const bf16_t* p = base + (size_t)n * rs + kb;
  BFrag f;
  f.q[0] = *(const uint4*)(p);
  f.q[1] = *(const uint4*)(p + 8);
  return f.v;
}

__device__ __forceinline__ v8f wmma_bf16(v16bf a, v16bf b, v8f c) {
  return __builtin_amdgcn_wmma_f32_16x16x32_bf16(false, a, false, b, (short)0, c,
                                                 false, false);
}

// ---------------------------------------------------------------------------
// Conversion / repack kernels
// ---------------------------------------------------------------------------
__global__ void cvt_bf16_kernel(const float* __restrict__ src,
                                bf16_t* __restrict__ dst, long n) {
  long i = (long)blockIdx.x * blockDim.x + threadIdx.x;
  long stride = (long)gridDim.x * blockDim.x;
  for (; i < n; i += stride) dst[i] = f2bf(src[i]);
}

// Wc [co][ci][3][3] (OIHW) -> wr [tap][co][ci] bf16, tap = dy*3+dx
__global__ void repack_conv_kernel(const float* __restrict__ W,
                                   bf16_t* __restrict__ dst) {
  int idx = blockIdx.x * 256 + threadIdx.x;          // < 9*256*256
  int tap = idx >> 16;
  int rem = idx & 65535;                              // co*256 + ci
  dst[idx] = f2bf(W[(size_t)rem * 9 + tap]);
}

// ---------------------------------------------------------------------------
// 256-K GEMM: out[m, col] = sum_k A[m,k] * W[col,k]
// MODE 0: bf16 out, [M][256] layout.   MODE 1: fp32 out + bias, [M][256].
// MODE 2: bf16 out, transposed [b][256 ch][16384 n] layout (for q/k).
// Block tile 128(M) x 64(N), 8 waves -> 32x32 per wave.
// Double-buffered LDS filled by async global->LDS (6 issues/wave/stage).
// ---------------------------------------------------------------------------
template <int MODE>
__global__ __launch_bounds__(256) void gemm256_kernel(
    const bf16_t* __restrict__ A, const bf16_t* __restrict__ W,
    const float* __restrict__ bias, void* __restrict__ outv) {
  __shared__ __align__(16) bf16_t As[2][128 * 64];
  __shared__ __align__(16) bf16_t Bs[2][64 * 64];
  const int m0 = blockIdx.x * 128;
  const int n0 = blockIdx.y * 64;
  const int t = threadIdx.x;
  const int w = t >> 5, lane = t & 31;
  const int wm = w & 3, wn = w >> 2;
  const int half = lane >> 4, col16 = lane & 15;
  const int rowA = t >> 1, partA = t & 1;
  const int rowB = t >> 2, partB = t & 3;
  const bf16_t* gA = A + (size_t)(m0 + rowA) * 256 + partA * 32;
  const bf16_t* gB = W + (size_t)(n0 + rowB) * 256 + partB * 16;
  v8f acc[2][2] = {};

  auto issue = [&](int buf, int kk) {
    unsigned la = lds_addr(&As[buf][rowA * 64 + partA * 32]);
    const bf16_t* ga = gA + kk;
#pragma unroll
    for (int j = 0; j < 4; ++j) async_ld16(la + j * 16, ga + j * 8);
    unsigned lb = lds_addr(&Bs[buf][rowB * 64 + partB * 16]);
    const bf16_t* gb = gB + kk;
    async_ld16(lb, gb);
    async_ld16(lb + 16, gb + 8);
  };

  issue(0, 0);
  for (int i = 0; i < 4; ++i) {
    if (i < 3) { issue((i + 1) & 1, (i + 1) * 64); wait_async<6>(); }
    else       { wait_async<0>(); }
    __syncthreads();
    const bf16_t* as = As[i & 1];
    const bf16_t* bs = Bs[i & 1];
#pragma unroll
    for (int ks = 0; ks < 64; ks += 32) {
      v16bf a0 = load_frag_a(as, 64, wm * 32, ks);
      v16bf a1 = load_frag_a(as, 64, wm * 32 + 16, ks);
      v16bf b0 = load_frag_b(bs, 64, wn * 32, ks);
      v16bf b1 = load_frag_b(bs, 64, wn * 32 + 16, ks);
      acc[0][0] = wmma_bf16(a0, b0, acc[0][0]);
      acc[0][1] = wmma_bf16(a0, b1, acc[0][1]);
      acc[1][0] = wmma_bf16(a1, b0, acc[1][0]);
      acc[1][1] = wmma_bf16(a1, b1, acc[1][1]);
    }
    __syncthreads();
  }

#pragma unroll
  for (int im = 0; im < 2; ++im)
#pragma unroll
    for (int ie = 0; ie < 2; ++ie) {
      int mbase = m0 + wm * 32 + im * 16 + half * 8;   // D row M = r + half*8
      int col = n0 + wn * 32 + ie * 16 + col16;        // D col N = lane&15
      if (MODE == 0) {
        bf16_t* out = (bf16_t*)outv;
#pragma unroll
        for (int r = 0; r < 8; ++r)
          out[(size_t)(mbase + r) * 256 + col] = f2bf(acc[im][ie][r]);
      } else if (MODE == 1) {
        float* out = (float*)outv;
        float bv = bias[col];
#pragma unroll
        for (int r = 0; r < 8; ++r)
          out[(size_t)(mbase + r) * 256 + col] = acc[im][ie][r] + bv;
      } else {  // MODE 2: transposed [b][c][n], 8 contiguous n -> one b128 store
        bf16_t* out = (bf16_t*)outv;
        int b = mbase >> 14, n = mbase & 16383;
        union { uint4 q; bf16_t h[8]; } pk;
#pragma unroll
        for (int r = 0; r < 8; ++r) pk.h[r] = f2bf(acc[im][ie][r]);
        *(uint4*)(out + ((size_t)(b * 256 + col) << 14) + n) = pk.q;
      }
    }
}

// ---------------------------------------------------------------------------
// Inverse L2 norms over n for each (b, channel) row of transposed q/k.
// ---------------------------------------------------------------------------
__global__ __launch_bounds__(256) void norms_kernel(
    const bf16_t* __restrict__ qT, const bf16_t* __restrict__ kT,
    float* __restrict__ invq, float* __restrict__ invk) {
  __shared__ float red[2][256];
  int row = blockIdx.x;  // 0..2047 = b*256 + c
  const bf16_t* q = qT + (size_t)row * 16384;
  const bf16_t* k = kT + (size_t)row * 16384;
  float sq = 0.f, sk = 0.f;
  for (int i = threadIdx.x; i < 16384; i += 256) {
    float a = bf2f(q[i]); sq += a * a;
    float b = bf2f(k[i]); sk += b * b;
  }
  int t = threadIdx.x;
  red[0][t] = sq; red[1][t] = sk;
  __syncthreads();
  for (int s = 128; s > 0; s >>= 1) {
    if (t < s) { red[0][t] += red[0][t + s]; red[1][t] += red[1][t + s]; }
    __syncthreads();
  }
  if (t == 0) {
    invq[row] = 1.f / fmaxf(sqrtf(red[0][0]), 1e-12f);
    invk[row] = 1.f / fmaxf(sqrtf(red[1][0]), 1e-12f);
  }
}

// ---------------------------------------------------------------------------
// Attention scores: per (b,h), S = K·Qᵀ over n (K=16384) via WMMA, then
// scale by invk[d]*invq[e]*rescale[h], softmax over e, store bf16 attn[32][32].
// ---------------------------------------------------------------------------
__global__ __launch_bounds__(256) void attn_scores_kernel(
    const bf16_t* __restrict__ qT, const bf16_t* __restrict__ kT,
    const float* __restrict__ invq, const float* __restrict__ invk,
    const float* __restrict__ rescale, bf16_t* __restrict__ attnw) {
  __shared__ float Sp[8][32][32];
  int bh = blockIdx.x;
  int b = bh >> 3, h = bh & 7;
  int t = threadIdx.x, w = t >> 5, lane = t & 31;
  int half = lane >> 4, col16 = lane & 15;
  const bf16_t* kbase = kT + ((size_t)(b * 256 + h * 32)) * 16384;
  const bf16_t* qbase = qT + ((size_t)(b * 256 + h * 32)) * 16384;
  v8f acc[2][2] = {};
  for (int nc = w; nc < 512; nc += 8) {  // 512 k-steps of 32 over n
    int k0 = nc * 32;
    v16bf a0 = load_frag_a(kbase, 16384, 0, k0);
    v16bf a1 = load_frag_a(kbase, 16384, 16, k0);
    v16bf b0 = load_frag_b(qbase, 16384, 0, k0);
    v16bf b1 = load_frag_b(qbase, 16384, 16, k0);
    acc[0][0] = wmma_bf16(a0, b0, acc[0][0]);
    acc[0][1] = wmma_bf16(a0, b1, acc[0][1]);
    acc[1][0] = wmma_bf16(a1, b0, acc[1][0]);
    acc[1][1] = wmma_bf16(a1, b1, acc[1][1]);
  }
#pragma unroll
  for (int im = 0; im < 2; ++im)
#pragma unroll
    for (int ie = 0; ie < 2; ++ie)
#pragma unroll
      for (int r = 0; r < 8; ++r)
        Sp[w][im * 16 + half * 8 + r][ie * 16 + col16] = acc[im][ie][r];
  __syncthreads();
  if (t < 32) {
    int d = t;
    float rowv[32];
    float sik = invk[b * 256 + h * 32 + d];
    float rs = rescale[h];
    float mx = -1e30f;
#pragma unroll
    for (int e = 0; e < 32; ++e) {
      float s = 0.f;
#pragma unroll
      for (int ww = 0; ww < 8; ++ww) s += Sp[ww][d][e];
      s = s * sik * invq[b * 256 + h * 32 + e] * rs;
      rowv[e] = s;
      mx = fmaxf(mx, s);
    }
    float den = 0.f;
#pragma unroll
    for (int e = 0; e < 32; ++e) { rowv[e] = __expf(rowv[e] - mx); den += rowv[e]; }
    float inv = 1.f / den;
#pragma unroll
    for (int e = 0; e < 32; ++e)
      attnw[bh * 1024 + d * 32 + e] = f2bf(rowv[e] * inv);
  }
}

// ---------------------------------------------------------------------------
// Attention apply: xo[d,n] = sum_e attn[d,e] * v[e,n]; two WMMAs (d 0..15, 16..31) per 16-n tile
// (K=32 covers all of e). v is NHWC so B-fragments are contiguous 32B loads.
// ---------------------------------------------------------------------------
__global__ __launch_bounds__(256) void attn_apply_kernel(
    const bf16_t* __restrict__ attnw, const bf16_t* __restrict__ vbf,
    bf16_t* __restrict__ xo) {
  __shared__ __align__(16) bf16_t Asm[32 * 32];
  int bh = blockIdx.x;
  int b = bh >> 3, h = bh & 7;
  int t = threadIdx.x, w = t >> 5, lane = t & 31;
  int half = lane >> 4, col16 = lane & 15;
  *(uint2*)(Asm + t * 4) = *(const uint2*)(attnw + bh * 1024 + t * 4);
  __syncthreads();
  v16bf a0 = load_frag_a(Asm, 32, 0, 0);    // d = 0..15
  v16bf a1 = load_frag_a(Asm, 32, 16, 0);   // d = 16..31
  int ncol = blockIdx.y * 128 + w * 16 + col16;
  const bf16_t* p = vbf + ((size_t)(b * 16384 + ncol)) * 256 + h * 32 + half * 16;
  BFrag bf;
  bf.q[0] = *(const uint4*)(p);
  bf.q[1] = *(const uint4*)(p + 8);
  v8f c0 = {}, c1 = {};
  c0 = wmma_bf16(a0, bf.v, c0);
  c1 = wmma_bf16(a1, bf.v, c1);
  bf16_t* out = xo + ((size_t)(b * 16384 + ncol)) * 256 + h * 32;
  union { uint4 q; bf16_t h8[8]; } pk;
#pragma unroll
  for (int r = 0; r < 8; ++r) pk.h8[r] = f2bf(c0[r]);
  *(uint4*)(out + half * 8) = pk.q;               // d = half*8 + r
#pragma unroll
  for (int r = 0; r < 8; ++r) pk.h8[r] = f2bf(c1[r]);
  *(uint4*)(out + 16 + half * 8) = pk.q;          // d = 16 + half*8 + r
}

// ---------------------------------------------------------------------------
// 3x3 conv (SAME, NHWC, 256->256) as implicit GEMM: K = 9 taps x 256 ci.
// Block = one image row (128 x-positions) x 64 output channels.
// Double-buffered LDS + async global->LDS pipeline over (tap, ci) stages.
// Every wave issues exactly 6 async ops per stage (boundary lanes zero-fill
// via ds stores), so the asynccnt<=6 pipeline threshold is uniform.
// GELU=1: exact gelu, bf16 out.  GELU=0: fp32 out.
// ---------------------------------------------------------------------------
template <int GELU>
__global__ __launch_bounds__(256) void conv3x3_kernel(
    const bf16_t* __restrict__ in, const bf16_t* __restrict__ wr,
    const float* __restrict__ bias, void* __restrict__ outv) {
  __shared__ __align__(16) bf16_t As[2][128 * 64];
  __shared__ __align__(16) bf16_t Bs[2][64 * 64];
  int by = blockIdx.x;
  int b = by >> 7, y = by & 127;
  int co0 = blockIdx.y * 64;
  int t = threadIdx.x, w = t >> 5, lane = t & 31;
  int half = lane >> 4, col16 = lane & 15;
  int wm = w & 3, wn = w >> 2;
  const int xr = t >> 1, partA = t & 1;
  const int rowB = t >> 2, partB = t & 3;
  v8f acc[2][2] = {};

  int tapok[9];
  int T = 0;
#pragma unroll
  for (int tap = 0; tap < 9; ++tap) {
    int yy = y + tap / 3 - 1;
    if (yy >= 0 && yy <= 127) tapok[T++] = tap;
  }
  const int S = T * 4;  // stages: valid taps x 4 ci-chunks of 64

  auto issue = [&](int buf, int s) {
    int tap = tapok[s >> 2];
    int ci0 = (s & 3) * 64;
    int dy = tap / 3 - 1, dx = tap % 3 - 1;
    int yy = y + dy;
    int xx = xr + dx;
    bf16_t* adst = &As[buf][xr * 64 + partA * 32];
    if (xx >= 0 && xx < 128) {
      const bf16_t* ga =
          in + ((size_t)((b * 128 + yy) * 128 + xx)) * 256 + ci0 + partA * 32;
      unsigned la = lds_addr(adst);
#pragma unroll
      for (int j = 0; j < 4; ++j) async_ld16(la + j * 16, ga + j * 8);
    } else {
      uint4 z = {0u, 0u, 0u, 0u};
      uint4* d = (uint4*)adst;
      d[0] = z; d[1] = z; d[2] = z; d[3] = z;
    }
    unsigned lb = lds_addr(&Bs[buf][rowB * 64 + partB * 16]);
    const bf16_t* gb =
        wr + ((size_t)(tap * 256 + co0 + rowB)) * 256 + ci0 + partB * 16;
    async_ld16(lb, gb);
    async_ld16(lb + 16, gb + 8);
  };

  issue(0, 0);
  for (int s = 0; s < S; ++s) {
    if (s + 1 < S) { issue((s + 1) & 1, s + 1); wait_async<6>(); }
    else           { wait_async<0>(); }
    __syncthreads();
    const bf16_t* as = As[s & 1];
    const bf16_t* bs = Bs[s & 1];
#pragma unroll
    for (int ks = 0; ks < 64; ks += 32) {
      v16bf a0 = load_frag_a(as, 64, wm * 32, ks);
      v16bf a1 = load_frag_a(as, 64, wm * 32 + 16, ks);
      v16bf b0 = load_frag_b(bs, 64, wn * 32, ks);
      v16bf b1 = load_frag_b(bs, 64, wn * 32 + 16, ks);
      acc[0][0] = wmma_bf16(a0, b0, acc[0][0]);
      acc[0][1] = wmma_bf16(a0, b1, acc[0][1]);
      acc[1][0] = wmma_bf16(a1, b0, acc[1][0]);
      acc[1][1] = wmma_bf16(a1, b1, acc[1][1]);
    }
    __syncthreads();
  }

#pragma unroll
  for (int im = 0; im < 2; ++im)
#pragma unroll
    for (int ie = 0; ie < 2; ++ie) {
      int co = co0 + wn * 32 + ie * 16 + col16;
      float bv = bias[co];
#pragma unroll
      for (int r = 0; r < 8; ++r) {
        int x = wm * 32 + im * 16 + half * 8 + r;
        float v = acc[im][ie][r] + bv;
        size_t idx = ((size_t)((b * 128 + y) * 128 + x)) * 256 + co;
        if (GELU) {
          v = 0.5f * v * (1.f + erff(v * 0.70710678118654752f));
          ((bf16_t*)outv)[idx] = f2bf(v);
        } else {
          ((float*)outv)[idx] = v;
        }
      }
    }
}

// ---------------------------------------------------------------------------
// Host-side orchestration
// ---------------------------------------------------------------------------
extern "C" void kernel_launch(void* const* d_in, const int* in_sizes, int n_in,
                              void* d_out, int out_size, void* d_ws, size_t ws_size,
                              hipStream_t stream) {
  (void)in_sizes; (void)n_in; (void)out_size; (void)ws_size;
  const float* x_in    = (const float*)d_in[0];
  const float* Wq      = (const float*)d_in[1];
  const float* Wk      = (const float*)d_in[2];
  const float* Wv      = (const float*)d_in[3];
  const float* rescale = (const float*)d_in[4];
  const float* Wp      = (const float*)d_in[5];
  const float* bp      = (const float*)d_in[6];
  const float* Wc1     = (const float*)d_in[7];
  const float* bc1     = (const float*)d_in[8];
  const float* Wc2     = (const float*)d_in[9];
  const float* bc2     = (const float*)d_in[10];

  const size_t NTOK = 8UL * 16384UL;  // 131072
  size_t off = 0;
  auto carve = [&](size_t bytes) -> char* {
    char* p = (char*)d_ws + off;
    off += (bytes + 255) & ~(size_t)255;
    return p;
  };
  bf16_t* xbf   = (bf16_t*)carve(NTOK * 256 * 2);
  bf16_t* qT    = (bf16_t*)carve(NTOK * 256 * 2);
  bf16_t* kT    = (bf16_t*)carve(NTOK * 256 * 2);
  bf16_t* vbf   = (bf16_t*)carve(NTOK * 256 * 2);
  bf16_t* xo    = (bf16_t*)carve(NTOK * 256 * 2);
  bf16_t* p1    = (bf16_t*)carve(NTOK * 256 * 2);
  bf16_t* wqb   = (bf16_t*)carve(65536 * 2);
  bf16_t* wkb   = (bf16_t*)carve(65536 * 2);
  bf16_t* wvb   = (bf16_t*)carve(65536 * 2);
  bf16_t* wpb   = (bf16_t*)carve(65536 * 2);
  bf16_t* wc1r  = (bf16_t*)carve(589824 * 2);
  bf16_t* wc2r  = (bf16_t*)carve(589824 * 2);
  float*  invq  = (float*)carve(2048 * 4);
  float*  invk  = (float*)carve(2048 * 4);
  bf16_t* attnw = (bf16_t*)carve(64 * 1024 * 2);

  // 1) convert inputs / weights to bf16, repack conv weights
  cvt_bf16_kernel<<<4096, 256, 0, stream>>>(x_in, xbf, (long)(NTOK * 256));
  cvt_bf16_kernel<<<64, 256, 0, stream>>>(Wq, wqb, 65536);
  cvt_bf16_kernel<<<64, 256, 0, stream>>>(Wk, wkb, 65536);
  cvt_bf16_kernel<<<64, 256, 0, stream>>>(Wv, wvb, 65536);
  cvt_bf16_kernel<<<64, 256, 0, stream>>>(Wp, wpb, 65536);
  repack_conv_kernel<<<2304, 256, 0, stream>>>(Wc1, wc1r);
  repack_conv_kernel<<<2304, 256, 0, stream>>>(Wc2, wc2r);

  // 2) projections: q/k transposed [b][c][n], v in NHWC
  dim3 ggrid(1024, 4);
  gemm256_kernel<2><<<ggrid, 256, 0, stream>>>(xbf, wqb, nullptr, qT);
  gemm256_kernel<2><<<ggrid, 256, 0, stream>>>(xbf, wkb, nullptr, kT);
  gemm256_kernel<0><<<ggrid, 256, 0, stream>>>(xbf, wvb, nullptr, vbf);

  // 3) attention
  norms_kernel<<<2048, 256, 0, stream>>>(qT, kT, invq, invk);
  attn_scores_kernel<<<64, 256, 0, stream>>>(qT, kT, invq, invk, rescale, attnw);
  attn_apply_kernel<<<dim3(64, 128), 256, 0, stream>>>(attnw, vbf, xo);

  // 4) output projection -> out_c (first half of d_out)
  gemm256_kernel<1><<<ggrid, 256, 0, stream>>>(xo, wpb, bp, d_out);

  // 5) conv branch -> out_p (second half of d_out)
  dim3 cgrid(1024, 4);
  conv3x3_kernel<1><<<cgrid, 256, 0, stream>>>(vbf, wc1r, bc1, p1);
  conv3x3_kernel<0><<<cgrid, 256, 0, stream>>>(p1, wc2r, bc2,
                                               (float*)d_out + 33554432);
}
